// OmniDirectionalTwoDimensionalGRU_57732950393179
// MI455X (gfx1250) — matrix operand order repaired and checked by reading, not compiled, not run on hardware
//
#include <hip/hip_runtime.h>
#include <hip/hip_bf16.h>

typedef __attribute__((ext_vector_type(16))) _Float16 v16h;
typedef __attribute__((ext_vector_type(8)))  _Float16 v8h;
typedef __attribute__((ext_vector_type(8)))  float    v8f;
typedef __attribute__((ext_vector_type(4)))  int      v4i;
typedef __attribute__((address_space(1))) v4i glob_v4i;
typedef __attribute__((address_space(3))) v4i lds_v4i;

#define WMMA16(a, b, c) __builtin_amdgcn_wmma_f32_16x16x32_f16( \
    false, (a), false, (b), (short)0, (c), false, false)

static constexpr int Bn   = 16;    // batch
static constexpr int PR   = 32, PC = 32;
static constexpr int INn  = 64;
static constexpr int En   = 256;   // embed dim
static constexpr int Hn   = 256;   // hidden per cell
static constexpr int Gn   = 512;   // 2H
static constexpr int Dn   = 4;     // directions
static constexpr int GK   = 768;   // fused K: [h_prev_row(256) | h_prev_col(256) | emb(256)]
static constexpr int GNg  = 2048;  // fused gate N: r(512) z(512) xn(512) hn(512)
static constexpr int CELL = Bn * Hn;  // halfs per h-grid cell block

// ---------------------------------------------------------------------------
// CDNA5 async global->LDS copy (ASYNCcnt path), with safe fallback.
// ---------------------------------------------------------------------------
#if defined(__HIP_DEVICE_COMPILE__) && __has_builtin(__builtin_amdgcn_global_load_async_to_lds_b128)
#define HAVE_ASYNC_LDS 1
#else
#define HAVE_ASYNC_LDS 0
#endif

template <int CPOL>
__device__ __forceinline__ void cp16_g2l(const _Float16* g, _Float16* l) {
#if HAVE_ASYNC_LDS
  __builtin_amdgcn_global_load_async_to_lds_b128((glob_v4i*)g, (lds_v4i*)l, 0, CPOL);
#else
  *(v8h*)l = *(const v8h*)g;
#endif
}

__device__ __forceinline__ void wait_async0() {
#if HAVE_ASYNC_LDS
#if __has_builtin(__builtin_amdgcn_s_wait_asynccnt)
  __builtin_amdgcn_s_wait_asynccnt(0);
#else
  asm volatile("s_wait_asynccnt 0x0" ::: "memory");
#endif
#endif
}

// ---------------------------------------------------------------------------
// Weight prep kernels (f32 -> f16, fused layouts)
// ---------------------------------------------------------------------------
__global__ void k_cvt_wemb(const float* __restrict__ w, _Float16* __restrict__ o, int n) {
  int i = blockIdx.x * blockDim.x + threadIdx.x;
  if (i < n) o[i] = (_Float16)w[i];
}

// Wg[d][g][k]: g<512: fused r = [Whh_r | Wih_r]; 512..1023: fused z;
// 1024..1535: xn = [0 | Wih_n]; 1536..2047: hn = [Whh_n | 0]
__global__ void k_build_wg(const float* __restrict__ Wih, const float* __restrict__ Whh,
                           _Float16* __restrict__ Wg) {
  int idx = blockIdx.x * blockDim.x + threadIdx.x;
  if (idx >= Dn * GNg * GK) return;
  int d = idx / (GNg * GK);
  int rem = idx % (GNg * GK);
  int g = rem / GK;
  int kk = rem % GK;
  float v;
  if (g < 1024) {
    v = (kk < 512) ? Whh[((size_t)d * 1536 + g) * 512 + kk]
                   : Wih[((size_t)d * 1536 + g) * 256 + (kk - 512)];
  } else if (g < 1536) {
    v = (kk < 512) ? 0.0f : Wih[((size_t)d * 1536 + g) * 256 + (kk - 512)];
  } else {
    v = (kk < 512) ? Whh[((size_t)d * 1536 + (g - 512)) * 512 + kk] : 0.0f;
  }
  Wg[idx] = (_Float16)v;
}

__global__ void k_build_bg(const float* __restrict__ bih, const float* __restrict__ bhh,
                           float* __restrict__ bg) {
  int idx = blockIdx.x * blockDim.x + threadIdx.x;
  if (idx >= Dn * GNg) return;
  int d = idx / GNg, g = idx % GNg;
  float v;
  if (g < 1024)       v = bih[d * 1536 + g] + bhh[d * 1536 + g];
  else if (g < 1536)  v = bih[d * 1536 + g];
  else                v = bhh[d * 1536 + (g - 512)];
  bg[idx] = v;
}

// Wo[d][h][k]: [Wc | Wr]
__global__ void k_build_wo(const float* __restrict__ Wc, const float* __restrict__ Wr,
                           _Float16* __restrict__ Wo) {
  int idx = blockIdx.x * blockDim.x + threadIdx.x;
  if (idx >= Dn * Hn * GK) return;
  int d = idx / (Hn * GK);
  int rem = idx % (Hn * GK);
  int h = rem / GK;
  int kk = rem % GK;
  float v = (kk < 512) ? Wc[((size_t)d * Hn + h) * 512 + kk]
                       : Wr[((size_t)d * Hn + h) * 256 + (kk - 512)];
  Wo[idx] = (_Float16)v;
}

__global__ void k_build_bo(const float* __restrict__ bc, const float* __restrict__ br,
                           float* __restrict__ bo) {
  int idx = blockIdx.x * blockDim.x + threadIdx.x;
  if (idx < Dn * Hn) bo[idx] = bc[idx] + br[idx];
}

__global__ void k_zero_u32(unsigned int* __restrict__ p, int n) {
  int i = blockIdx.x * blockDim.x + threadIdx.x;
  if (i < n) p[i] = 0u;
}

// ---------------------------------------------------------------------------
// Embedding GEMM: emb[b,r,c,:] = x[b,r,c,:] @ Wemb^T + bemb  (M=16384,K=64,N=256)
// One wave per 16x16 output tile.
// ---------------------------------------------------------------------------
__global__ __launch_bounds__(256) void k_embed(const float* __restrict__ x,
                                               const _Float16* __restrict__ WembH,
                                               const float* __restrict__ bemb,
                                               _Float16* __restrict__ emb) {
  int tid = threadIdx.x;
  int wid = blockIdx.x * (blockDim.x >> 5) + (tid >> 5);
  int lane = tid & 31;
  int ncol = lane & 15, hi = lane >> 4;
  int mtile = wid >> 4;   // 1024
  int ntile = wid & 15;   // 16
  int arow = mtile * 16 + ncol;

  v8f acc;
#pragma unroll
  for (int v = 0; v < 8; ++v) acc[v] = 0.0f;

#pragma unroll
  for (int ks = 0; ks < 2; ++ks) {
    const float* xr = x + (size_t)arow * INn + ks * 32;
    v16h a;
#pragma unroll
    for (int i = 0; i < 8; ++i) a[i] = (_Float16)xr[hi * 16 + i];
#pragma unroll
    for (int i = 0; i < 8; ++i) a[8 + i] = (_Float16)xr[hi * 16 + 8 + i];
    v16h b = *(const v16h*)(WembH + ((size_t)(ntile * 16 + ncol)) * INn + ks * 32 + hi * 16);
    acc = WMMA16(a, b, acc);
  }
  float bias = bemb[ntile * 16 + ncol];
#pragma unroll
  for (int v = 0; v < 8; ++v) {
    int m = hi * 8 + v;
    emb[((size_t)(mtile * 16 + m)) * En + ntile * 16 + ncol] = (_Float16)(acc[v] + bias);
  }
}

// ---------------------------------------------------------------------------
// Persistent wavefront kernel: WG (r,d) owns row r of direction d and walks
// columns 0..31. Prev-col state stays in LDS (Hcol); prev-row state arrives
// via per-row progress flags + async global->LDS loads.
// ---------------------------------------------------------------------------
__global__ __launch_bounds__(512) void k_rows(const _Float16* __restrict__ emb,
                                              const _Float16* __restrict__ Wg,
                                              const float* __restrict__ bg,
                                              const _Float16* __restrict__ Wo,
                                              const float* __restrict__ bo,
                                              const float* __restrict__ gamma,
                                              const float* __restrict__ beta,
                                              _Float16* __restrict__ hgrid,
                                              float* __restrict__ out,
                                              int* __restrict__ progress) {
  __shared__ __align__(32) _Float16 Aprev[16 * 256];  // h(r-1,c)
  __shared__ __align__(32) _Float16 Hcol[16 * 256];   // h(r,c-1)
  __shared__ __align__(32) _Float16 Aemb[16 * 256];   // emb(flipped r,c)
  __shared__ __align__(32) _Float16 Hnew[16 * 512];   // GRU hidden (2H)
  __shared__ float s1[16], s2[16];

  const int r = blockIdx.x;   // row in scan coords
  const int d = blockIdx.y;   // direction
  const int tid = threadIdx.x;
  const int w = tid >> 5, lane = tid & 31;
  const int ncol = lane & 15, hi = lane >> 4;
  const int am = tid >> 5, acol = (tid & 31) * 8;  // one 16B chunk per thread

  for (int i = tid; i < 16 * 256; i += 512) {
    Hcol[i] = (_Float16)0.0f;
    if (r == 0) Aprev[i] = (_Float16)0.0f;
  }

  int* prog = progress + d * PR;
  const _Float16* Wgd = Wg + (size_t)d * GNg * GK;
  const _Float16* Wod = Wo + (size_t)d * Hn * GK;
  const float* bgd = bg + d * GNg;
  const float* bod = bo + d * Hn;
  const int rs = (d & 2) ? (PR - 1 - r) : r;
  const float ga = gamma[d * Hn + w * 16 + ncol];
  const float be = beta[d * Hn + w * 16 + ncol];

  for (int c = 0; c < PC; ++c) {
    const int cs = (d & 1) ? (PC - 1 - c) : c;

    // issue emb tile fetch early (no cross-WG dependency)
    cp16_g2l<0>(emb + (((size_t)am * PR + rs) * PC + cs) * En + acol, &Aemb[am * En + acol]);

    if (r > 0) {
      if (tid == 0) {
        while (__hip_atomic_load(&prog[r - 1], __ATOMIC_ACQUIRE, __HIP_MEMORY_SCOPE_AGENT) <= c)
          __builtin_amdgcn_s_sleep(1);
      }
      __syncthreads();
      __threadfence();
      cp16_g2l<0x10>(hgrid + (((size_t)d * PR + (r - 1)) * PC + c) * CELL + am * Hn + acol,
                     &Aprev[am * Hn + acol]);
    }
    wait_async0();
    __syncthreads();

    // ---- phase 1: gates(16x2048) = [hprev_row|hprev_col|emb] @ Wg^T ----
    v8f acc[8];
#pragma unroll
    for (int j = 0; j < 2; ++j)
#pragma unroll
      for (int p = 0; p < 4; ++p) {
        float bv = bgd[(2 * w + j + 32 * p) * 16 + ncol];
        v8f t;
#pragma unroll
        for (int v = 0; v < 8; ++v) t[v] = bv;
        acc[j * 4 + p] = t;
      }
    for (int ks = 0; ks < GK / 32; ++ks) {
      const _Float16* segbase = (ks < 8) ? Aprev : (ks < 16) ? Hcol : Aemb;
      const _Float16* ar = segbase + ncol * En + (ks & 7) * 32;
      v8h lo8 = *(const v8h*)(ar + hi * 16);
      v8h hi8 = *(const v8h*)(ar + hi * 16 + 8);
      v16h a;
#pragma unroll
      for (int i = 0; i < 8; ++i) { a[i] = lo8[i]; a[8 + i] = hi8[i]; }
#pragma unroll
      for (int j = 0; j < 2; ++j)
#pragma unroll
        for (int p = 0; p < 4; ++p) {
          int nt = 2 * w + j + 32 * p;
          v16h b = *(const v16h*)(Wgd + ((size_t)(nt * 16 + ncol)) * GK + ks * 32 + hi * 16);
          acc[j * 4 + p] = WMMA16(a, b, acc[j * 4 + p]);
        }
    }

    // ---- GRU elementwise -> Hnew (separate buffer, no barrier needed) ----
#pragma unroll
    for (int j = 0; j < 2; ++j) {
      int t = 2 * w + j;
      const _Float16* hpb = (t < 16) ? (Aprev + t * 16 + ncol) : (Hcol + (t - 16) * 16 + ncol);
#pragma unroll
      for (int v = 0; v < 8; ++v) {
        int m = hi * 8 + v;
        float rr = 1.0f / (1.0f + __expf(-acc[j * 4 + 0][v]));
        float zz = 1.0f / (1.0f + __expf(-acc[j * 4 + 1][v]));
        float nn = tanhf(acc[j * 4 + 2][v] + rr * acc[j * 4 + 3][v]);
        float hp = (float)hpb[m * En];
        Hnew[m * Gn + t * 16 + ncol] = (_Float16)((1.0f - zz) * nn + zz * hp);
      }
    }
    if (tid < 16) { s1[tid] = 0.0f; s2[tid] = 0.0f; }
    __syncthreads();

    // ---- phase 2: y(16x256) = [hnew|emb] @ [Wc|Wr]^T + bias ----
    v8f acc2;
    {
      float bv = bod[w * 16 + ncol];
#pragma unroll
      for (int v = 0; v < 8; ++v) acc2[v] = bv;
    }
    for (int ks = 0; ks < GK / 32; ++ks) {
      const _Float16* ar = (ks < 16) ? (Hnew + ncol * Gn + ks * 32)
                                     : (Aemb + ncol * En + (ks - 16) * 32);
      v8h lo8 = *(const v8h*)(ar + hi * 16);
      v8h hi8 = *(const v8h*)(ar + hi * 16 + 8);
      v16h a;
#pragma unroll
      for (int i = 0; i < 8; ++i) { a[i] = lo8[i]; a[8 + i] = hi8[i]; }
      v16h b = *(const v16h*)(Wod + ((size_t)(w * 16 + ncol)) * GK + ks * 32 + hi * 16);
      acc2 = WMMA16(a, b, acc2);
    }

    // ---- phase 3: LayerNorm over H=256 per batch row ----
    v8f red = acc2, redsq;
#pragma unroll
    for (int v = 0; v < 8; ++v) redsq[v] = acc2[v] * acc2[v];
    for (int off = 1; off < 16; off <<= 1) {
#pragma unroll
      for (int v = 0; v < 8; ++v) {
        red[v] += __shfl_xor(red[v], off, 32);
        redsq[v] += __shfl_xor(redsq[v], off, 32);
      }
    }
    if (ncol == 0) {
#pragma unroll
      for (int v = 0; v < 8; ++v) {
        atomicAdd(&s1[hi * 8 + v], red[v]);
        atomicAdd(&s2[hi * 8 + v], redsq[v]);
      }
    }
    __syncthreads();

    int col = w * 16 + ncol;
    _Float16* hdst = hgrid + (((size_t)d * PR + r) * PC + c) * CELL;
    bool corner = (r == PR - 1 && c == PC - 1);
#pragma unroll
    for (int v = 0; v < 8; ++v) {
      int m = hi * 8 + v;
      float mu = s1[m] * (1.0f / 256.0f);
      float var = s2[m] * (1.0f / 256.0f) - mu * mu;
      float rstd = rsqrtf(var + 1e-5f);
      float hv = (acc2[v] - mu) * rstd * ga + be;
      hdst[m * Hn + col] = (_Float16)hv;
      Hcol[m * En + col] = (_Float16)hv;
      if (corner) out[(size_t)m * (Dn * Hn) + d * Hn + col] = hv;
    }
    __threadfence();
    __syncthreads();
    if (tid == 0)
      __hip_atomic_store(&prog[r], c + 1, __ATOMIC_RELEASE, __HIP_MEMORY_SCOPE_AGENT);
  }
}

// ---------------------------------------------------------------------------
extern "C" void kernel_launch(void* const* d_in, const int* in_sizes, int n_in,
                              void* d_out, int out_size, void* d_ws, size_t ws_size,
                              hipStream_t stream) {
  const float* x     = (const float*)d_in[0];
  const float* Wemb  = (const float*)d_in[1];
  const float* bemb  = (const float*)d_in[2];
  const float* Wih   = (const float*)d_in[3];
  const float* Whh   = (const float*)d_in[4];
  const float* bih   = (const float*)d_in[5];
  const float* bhh   = (const float*)d_in[6];
  const float* Wc    = (const float*)d_in[7];
  const float* bc    = (const float*)d_in[8];
  const float* Wr    = (const float*)d_in[9];
  const float* br    = (const float*)d_in[10];
  const float* gamma = (const float*)d_in[11];
  const float* beta  = (const float*)d_in[12];
  float* out = (float*)d_out;

  char* p = (char*)d_ws;
  auto carve = [&](size_t bytes) {
    char* q = p;
    p += (bytes + 255) & ~(size_t)255;
    return q;
  };
  _Float16* embH  = (_Float16*)carve((size_t)Bn * PR * PC * En * 2);    // 8 MB
  _Float16* WembH = (_Float16*)carve((size_t)En * INn * 2);
  _Float16* WgH   = (_Float16*)carve((size_t)Dn * GNg * GK * 2);        // 12.6 MB
  float*    bg    = (float*)carve((size_t)Dn * GNg * 4);
  _Float16* WoH   = (_Float16*)carve((size_t)Dn * Hn * GK * 2);         // 1.6 MB
  float*    bo    = (float*)carve((size_t)Dn * Hn * 4);
  _Float16* hgrid = (_Float16*)carve((size_t)Dn * PR * PC * CELL * 2);  // 33.5 MB
  int*      prog  = (int*)carve((size_t)Dn * PR * 4);

  const int T = 256;
  k_zero_u32<<<1, T, 0, stream>>>((unsigned int*)prog, Dn * PR);
  k_cvt_wemb<<<(En * INn + T - 1) / T, T, 0, stream>>>(Wemb, WembH, En * INn);
  k_build_wg<<<(Dn * GNg * GK + T - 1) / T, T, 0, stream>>>(Wih, Whh, WgH);
  k_build_bg<<<(Dn * GNg + T - 1) / T, T, 0, stream>>>(bih, bhh, bg);
  k_build_wo<<<(Dn * Hn * GK + T - 1) / T, T, 0, stream>>>(Wc, Wr, WoH);
  k_build_bo<<<(Dn * Hn + T - 1) / T, T, 0, stream>>>(bc, br, bo);

  // embedding GEMM: 1024 mtiles x 16 ntiles, 8 waves/block
  k_embed<<<(Bn * PR * PC / 16) * 16 / 8, T, 0, stream>>>(x, WembH, bemb, embH);

  // persistent pipelined wavefront: one WG per (row, direction)
  k_rows<<<dim3(PR, Dn), 512, 0, stream>>>(embH, WgH, bg, WoH, bo,
                                           gamma, beta, hgrid, out, prog);
}
